// Head_7284264534628
// MI455X (gfx1250) — matrix operand order repaired and checked by reading, not compiled, not run on hardware
//
#include <hip/hip_runtime.h>

typedef _Float16 f16;
typedef __attribute__((ext_vector_type(16))) _Float16 v16h;
typedef __attribute__((ext_vector_type(8)))  _Float16 v8h;
typedef __attribute__((ext_vector_type(8)))  float    v8f;

#define B_  16
#define T_  4096
#define C_  64
#define H_  128
#define SCALE 0.125f   // C_**-0.5

// ---- Tensor Data Mover support (probe: 5-arg on clang-22, 6-arg on 23+) ----
#if defined(__has_builtin)
# if __has_builtin(__builtin_amdgcn_tensor_load_to_lds) && \
     __has_builtin(__builtin_amdgcn_s_wait_tensorcnt)
#  define HAVE_TDM 1
# endif
#endif
#ifndef HAVE_TDM
# define HAVE_TDM 0
#endif

#if HAVE_TDM
typedef unsigned int v4u __attribute__((ext_vector_type(4)));
typedef int          v8i __attribute__((ext_vector_type(8)));
typedef int          v4i __attribute__((ext_vector_type(4)));

// 2D tile load: tensor [d1 rows][d0 elems] (f16), row stride stride0 elems,
// tile [tile_d1 rows][tile_d0 elems] starting at gsrc, packed into LDS.
__device__ __forceinline__ void tdm_load_2d(const f16* gsrc, unsigned lds_off,
                                            unsigned d0, unsigned d1,
                                            unsigned tile_d0, unsigned tile_d1,
                                            unsigned stride0) {
  unsigned long long ga = (unsigned long long)(size_t)gsrc;
  v4u g0 = { 1u,                                   // count=1, user mode
             lds_off,                              // lds_addr
             (unsigned)ga,                         // global_addr[31:0]
             (unsigned)(ga >> 32) | (2u << 30) };  // global_addr[56:32] | type=2
  v8i g1 = { (int)(1u << 16),                                          // data_size=2B
             (int)((d0 & 0xFFFFu) << 16),                              // tensor_dim0 lo
             (int)(((d0 >> 16) & 0xFFFFu) | ((d1 & 0xFFFFu) << 16)),   // d0 hi | d1 lo
             (int)(((d1 >> 16) & 0xFFFFu) | (tile_d0 << 16)),          // d1 hi | tile0
             (int)(tile_d1),                                           // tile1 | tile2=0
             (int)(stride0),                                           // dim0_stride lo
             0, 0 };
  v4i gz = {0, 0, 0, 0};
#if __clang_major__ >= 23
  v8i gz8 = {0, 0, 0, 0, 0, 0, 0, 0};
  __builtin_amdgcn_tensor_load_to_lds(g0, g1, gz, gz, gz8, 0);
#else
  __builtin_amdgcn_tensor_load_to_lds(g0, g1, gz, gz, 0);
#endif
}
#endif

// ---------------------------------------------------------------------------
// Kernel 0: convert weights f32 -> f16, transposed to [m][h][c] so that
// projection B-fragments are contiguous 32B loads.  m order: Q, K, V.
// ---------------------------------------------------------------------------
__global__ __launch_bounds__(256)
void wconv(const float* __restrict__ Wq, const float* __restrict__ Wk,
           const float* __restrict__ Wv, f16* __restrict__ Wt) {
  int idx = blockIdx.x * 256 + threadIdx.x;       // 3*128*64 = 24576 total
  if (idx >= 3 * H_ * C_) return;
  int m = idx / (H_ * C_);
  int r = idx % (H_ * C_);
  int h = r / C_, c = r % C_;
  const float* W = (m == 0) ? Wq : (m == 1) ? Wk : Wv;
  Wt[idx] = (f16)W[c * H_ + h];
}

// ---------------------------------------------------------------------------
// Kernel 1: QKV projection with WMMA. One wave per 16-row tile of x.
// Q stored pre-scaled by C^-0.5 (folds softmax scale into the S GEMM).
// Q,K row-major f16 [bt][H]; V transposed f16 [b][h][t].
// ---------------------------------------------------------------------------
__global__ __launch_bounds__(32)
void qkv_proj(const float* __restrict__ x, const f16* __restrict__ Wt,
              f16* __restrict__ Qg, f16* __restrict__ Kg, f16* __restrict__ Vt) {
  const int lane = threadIdx.x;
  const int half = lane >> 4;
  const int ln   = lane & 15;
  const long long bt0 = (long long)blockIdx.x * 16;

  // A fragments (x tile 16x64 -> two 16x32 f16), lane = row, vectorized loads
  const float4* xr4 = (const float4*)(x + (bt0 + ln) * C_);
  float4 q0 = xr4[ 0 + 4 * half], q1 = xr4[ 1 + 4 * half];
  float4 q2 = xr4[ 2 + 4 * half], q3 = xr4[ 3 + 4 * half];
  float4 q4 = xr4[ 8 + 4 * half], q5 = xr4[ 9 + 4 * half];
  float4 q6 = xr4[10 + 4 * half], q7 = xr4[11 + 4 * half];
  float t0[16] = {q0.x,q0.y,q0.z,q0.w, q1.x,q1.y,q1.z,q1.w,
                  q2.x,q2.y,q2.z,q2.w, q3.x,q3.y,q3.z,q3.w};
  float t1[16] = {q4.x,q4.y,q4.z,q4.w, q5.x,q5.y,q5.z,q5.w,
                  q6.x,q6.y,q6.z,q6.w, q7.x,q7.y,q7.z,q7.w};
  v16h a0, a1;
  for (int i = 0; i < 16; ++i) { a0[i] = (f16)t0[i]; a1[i] = (f16)t1[i]; }

  for (int m = 0; m < 3; ++m) {
    const f16* Wm = Wt + m * (H_ * C_);
    for (int nt = 0; nt < 8; ++nt) {
      const int n0 = nt * 16;
      // B fragment b[i] = W[k = 16*half+i][n0+ln] = Wm[(n0+ln)*64 + k]  (contig)
      v16h b0 = *(const v16h*)&Wm[(n0 + ln) * C_ +      16 * half];
      v16h b1 = *(const v16h*)&Wm[(n0 + ln) * C_ + 32 + 16 * half];
      v8f c;
      for (int r = 0; r < 8; ++r) c[r] = 0.0f;
      c = __builtin_amdgcn_wmma_f32_16x16x32_f16(false, a0, false, b0,
                                                 (short)0, c, false, false);
      c = __builtin_amdgcn_wmma_f32_16x16x32_f16(false, a1, false, b1,
                                                 (short)0, c, false, false);
      if (m == 0) {          // Q, pre-scaled
        for (int r = 0; r < 8; ++r)
          Qg[(bt0 + r + 8 * half) * H_ + n0 + ln] = (f16)(c[r] * SCALE);
      } else if (m == 1) {   // K row-major
        for (int r = 0; r < 8; ++r)
          Kg[(bt0 + r + 8 * half) * H_ + n0 + ln] = (f16)c[r];
      } else {               // V transposed [b][h][t]
        const long long bb = bt0 >> 12;
        const long long tt = bt0 & (T_ - 1);
        for (int r = 0; r < 8; ++r)
          Vt[bb * (long long)H_ * T_ + (long long)(n0 + ln) * T_ +
             tt + r + 8 * half] = (f16)c[r];
      }
    }
  }
}

// ---------------------------------------------------------------------------
// Kernel 2: flash attention, 4 waves/block (wave = 16 query rows).
// K/V tiles staged by the Tensor Data Mover with double buffering
// (TENSORcnt), 32 WMMAs per 64-key chunk per wave.
// ---------------------------------------------------------------------------
__global__ __launch_bounds__(128)
void flash_attn(const f16* __restrict__ Qg, const f16* __restrict__ Kg,
                const f16* __restrict__ Vt, float* __restrict__ out) {
  __shared__ __align__(32) f16  Ks[2][64][H_];        // 32 KB, double buffered
  __shared__ __align__(32) f16  Vs[2][H_][64];        // 32 KB, double buffered
  __shared__ float Ssc[4][16][64];                    // 16 KB per-wave S
  __shared__ __align__(32) f16  Ps2[4][2][16][2][16]; //  8 KB, A-frag swizzled
  __shared__ float astat[4][16];

  const int tid  = threadIdx.x;
  const int wave = tid >> 5;
  const int lane = tid & 31;
  const int half = lane >> 4;
  const int ln   = lane & 15;
  const int b    = blockIdx.x >> 6;
  const int qt   = blockIdx.x & 63;
  const long long qrow0 = (long long)b * T_ + qt * 64 + wave * 16;

  // Q A-fragments, loaded once as 16B vectors (Q already carries the scale)
  v16h qa[4];
  const f16* qp = Qg + (qrow0 + ln) * H_;
  for (int kc = 0; kc < 4; ++kc) {
    v8h lo = *(const v8h*)(qp + kc * 32 +     16 * half);
    v8h hi = *(const v8h*)(qp + kc * 32 + 8 + 16 * half);
    for (int i = 0; i < 8; ++i) { qa[kc][i] = lo[i]; qa[kc][8 + i] = hi[i]; }
  }

  v8f o[8];
  for (int nt = 0; nt < 8; ++nt)
    for (int r = 0; r < 8; ++r) o[nt][r] = 0.0f;
  float m_row = -1e30f, l_row = 0.0f;                 // lanes 0..15

  const f16* Kb = Kg + (long long)b * T_ * H_;
  const f16* Vb = Vt + (long long)b * H_ * T_;

#if HAVE_TDM
  // prologue: DMA first chunk into buffer 0 (one issuing wave per tile)
  if (wave == 0)
    tdm_load_2d(Kb, (unsigned)(size_t)&Ks[0][0][0], H_, T_, H_, 64, H_);
  else if (wave == 1)
    tdm_load_2d(Vb, (unsigned)(size_t)&Vs[0][0][0], T_, H_, 64, H_, T_);
#endif

  for (int j = 0; j < T_; j += 64) {
    const int p = (j >> 6) & 1;
    __syncthreads();                                  // buffers/scratch reusable
#if HAVE_TDM
    if (j + 64 < T_) {
      if (wave == 0)
        tdm_load_2d(Kb + (long long)(j + 64) * H_,
                    (unsigned)(size_t)&Ks[p ^ 1][0][0], H_, T_, H_, 64, H_);
      else if (wave == 1)
        tdm_load_2d(Vb + (j + 64),
                    (unsigned)(size_t)&Vs[p ^ 1][0][0], T_, H_, 64, H_, T_);
      __builtin_amdgcn_s_wait_tensorcnt(1);           // chunk j complete
    } else {
      __builtin_amdgcn_s_wait_tensorcnt(0);
    }
#else
    {
      const uint4* ks = (const uint4*)(Kb + (long long)j * H_);
      uint4* kd = (uint4*)&Ks[p][0][0];
      for (int t = tid; t < 1024; t += 128) kd[t] = ks[t];
      const uint4* vsrc = (const uint4*)(Vb + (long long)tid * T_ + j);
      uint4* vdst = (uint4*)&Vs[p][tid][0];
      for (int t = 0; t < 8; ++t) vdst[t] = vsrc[t];
    }
#endif
    __syncthreads();                                  // tile visible to all

    // --- S = q k^T : 4 tiles x 4 WMMAs (K-dim 128) ---
    v8f st[4];
    for (int nt = 0; nt < 4; ++nt) {
      v8f s;
      for (int r = 0; r < 8; ++r) s[r] = 0.0f;
      for (int kc = 0; kc < 4; ++kc) {
        v16h bb = *(const v16h*)&Ks[p][nt * 16 + ln][kc * 32 + 16 * half];
        s = __builtin_amdgcn_wmma_f32_16x16x32_f16(false, qa[kc], false, bb,
                                                   (short)0, s, false, false);
      }
      st[nt] = s;
    }
    for (int nt = 0; nt < 4; ++nt)
      for (int r = 0; r < 8; ++r)
        Ssc[wave][r + 8 * half][nt * 16 + ln] = st[nt][r];
    __syncthreads();

    // --- online softmax: lane ln<16 owns row ln; P written pre-swizzled ---
    if (lane < 16) {
      float mx = m_row;
      for (int c = 0; c < 64; ++c) {
        float v = Ssc[wave][ln][c];
        mx = v > mx ? v : mx;
      }
      float sum = 0.0f;
      for (int c = 0; c < 64; ++c) {
        float pv = __expf(Ssc[wave][ln][c] - mx);
        // A-fragment swizzle: element (kc=c>>5, i=c&15, half=(c>>4)&1)
        Ps2[wave][(c >> 4) & 1][ln][c >> 5][c & 15] =
            (f16)pv;
        sum += pv;
      }
      float alpha = __expf(m_row - mx);
      l_row = l_row * alpha + sum;
      m_row = mx;
      astat[wave][ln] = alpha;
    }
    __syncthreads();

    // --- O = O*alpha + P V : 8 tiles x 2 WMMAs ---
    float al[8];
    for (int r = 0; r < 8; ++r) al[r] = astat[wave][r + 8 * half];
    v16h pa[2];
    for (int kc = 0; kc < 2; ++kc)
      pa[kc] = *(const v16h*)&Ps2[wave][half][ln][kc][0];
    for (int nt = 0; nt < 8; ++nt) {
      for (int r = 0; r < 8; ++r) o[nt][r] *= al[r];
      for (int kc = 0; kc < 2; ++kc) {
        v16h vb = *(const v16h*)&Vs[p][nt * 16 + ln][kc * 32 + 16 * half];
        o[nt] = __builtin_amdgcn_wmma_f32_16x16x32_f16(false, pa[kc], false,
                                                       vb, (short)0, o[nt],
                                                       false, false);
      }
    }
  }

  // --- epilogue: normalize by row sums, store f32 ---
  __syncthreads();
  if (lane < 16) astat[wave][ln] = 1.0f / l_row;
  __syncthreads();
  float inv[8];
  for (int r = 0; r < 8; ++r) inv[r] = astat[wave][r + 8 * half];
  for (int nt = 0; nt < 8; ++nt)
    for (int r = 0; r < 8; ++r)
      out[(qrow0 + r + 8 * half) * H_ + nt * 16 + ln] = o[nt][r] * inv[r];
}

// ---------------------------------------------------------------------------
extern "C" void kernel_launch(void* const* d_in, const int* in_sizes, int n_in,
                              void* d_out, int out_size, void* d_ws,
                              size_t ws_size, hipStream_t stream) {
  const float* x  = (const float*)d_in[0];
  const float* Wk = (const float*)d_in[1];
  const float* Wq = (const float*)d_in[2];
  const float* Wv = (const float*)d_in[3];

  const size_t BT = (size_t)B_ * T_;
  f16* Qg = (f16*)d_ws;
  f16* Kg = Qg + BT * H_;
  f16* Vt = Kg + BT * H_;
  f16* Wt = Vt + BT * H_;                 // 3*128*64 f16
  float* out = (float*)d_out;

  wconv<<<96, 256, 0, stream>>>(Wq, Wk, Wv, Wt);
  qkv_proj<<<(int)(BT / 16), 32, 0, stream>>>(x, Wt, Qg, Kg, Vt);
  flash_attn<<<B_ * (T_ / 64), 128, 0, stream>>>(Qg, Kg, Vt, out);
}
